// CCN2_63299228009053
// MI455X (gfx1250) — 16-bit and 8-bit matrix operand order repaired and checked by reading, compiled, not run on hardware
//
#include <hip/hip_runtime.h>

typedef __attribute__((ext_vector_type(16))) _Float16 v16h;
typedef __attribute__((ext_vector_type(8)))  float    v8f;
typedef __attribute__((ext_vector_type(8)))  int      v8i;
typedef int gv4i __attribute__((vector_size(16)));
typedef __attribute__((address_space(3))) gv4i* lds_v4i_ptr;

#define BATCH 16
#define NN    1025
#define NP    1088
#define NT16  (NP / 16)
#define NT32  (NP / 32)
#define DIM   128
#define THRESH2 0.003025f

#if __has_builtin(__builtin_amdgcn_global_load_async_to_lds_b128)
#define ASYNC_LDS 1
#endif

__global__ __launch_bounds__(256) void adj_kernel(
    const float* __restrict__ loc, const float* __restrict__ depot,
    unsigned char* __restrict__ A) {
  const int idx = blockIdx.x;
  const int b = idx / NP, i = idx % NP;
  float xi = 0.f, yi = 0.f;
  const bool vi = (i < NN);
  if (vi) {
    if (i == 0) { xi = depot[2 * b];  yi = depot[2 * b + 1]; }
    else { xi = loc[((size_t)b * 1024 + i - 1) * 2];
           yi = loc[((size_t)b * 1024 + i - 1) * 2 + 1]; }
  }
  unsigned char* row = A + (size_t)b * NP * NP + (size_t)i * NP;
  for (int j = threadIdx.x; j < NP; j += blockDim.x) {
    unsigned char v = 0;
    if (vi && j < NN) {
      float xj, yj;
      if (j == 0) { xj = depot[2 * b]; yj = depot[2 * b + 1]; }
      else { xj = loc[((size_t)b * 1024 + j - 1) * 2];
             yj = loc[((size_t)b * 1024 + j - 1) * 2 + 1]; }
      const float dx = xi - xj, dy = yi - yj;
      v = (dx * dx + dy * dy <= THRESH2) ? 1 : 0;
    }
    row[j] = v;
  }
}

__global__ __launch_bounds__(128) void fv0_kernel(
    const float* __restrict__ loc, const float* __restrict__ deadline,
    const float* __restrict__ depot, const float* __restrict__ W0,
    const float* __restrict__ b0, _Float16* __restrict__ fv0h) {
  const int d = threadIdx.x;
  const int idx = blockIdx.x;
  const int b = idx / NP, node = idx % NP;
  float v = 0.f;
  if (node < NN) {
    float lx, ly, td;
    if (node == 0) { lx = depot[2 * b]; ly = depot[2 * b + 1]; td = 0.f; }
    else {
      const size_t o = ((size_t)b * 1024 + node - 1);
      lx = loc[o * 2]; ly = loc[o * 2 + 1]; td = deadline[o];
    }
    v = fmaxf(lx * W0[d] + ly * W0[DIM + d] + td * W0[2 * DIM + d] + b0[d], 0.f);
  }
  fv0h[(size_t)idx * DIM + d] = (_Float16)v;
}

__global__ void cvt_kernel(const float* __restrict__ W2, _Float16* __restrict__ W2h) {
  const int i = blockIdx.x * 256 + threadIdx.x;
  if (i < DIM * DIM) W2h[i] = (_Float16)W2[i];
}

__global__ __launch_bounds__(256) void iu8_gemm_kernel(
    const unsigned char* __restrict__ Am, const unsigned char* __restrict__ Bm,
    const unsigned char* __restrict__ mask, unsigned char* __restrict__ Out,
    int mode) {
  const int wave = threadIdx.x >> 5;
  const int lane = threadIdx.x & 31;
  const int batch = blockIdx.y;
  const int tile = blockIdx.x * 8 + wave;
  if (tile >= NT32 * NT32) return;
  const size_t base = (size_t)batch * NP * NP;
  const int tmm = (tile / NT32) * 32;
  const int tnn = (tile % NT32) * 32;
  const int hi   = lane >> 4;
  const int mrow = lane & 15;
  const unsigned char* arow0 = Am + base + (size_t)(tmm + mrow) * NP;
  const unsigned char* arow1 = arow0 + (size_t)16 * NP;
  const unsigned char* brow0 = Bm + base + (size_t)(tnn + mrow) * NP;
  const unsigned char* brow1 = brow0 + (size_t)16 * NP;
  v8i acc00 = {0,0,0,0,0,0,0,0}, acc01 = {0,0,0,0,0,0,0,0};
  v8i acc10 = {0,0,0,0,0,0,0,0}, acc11 = {0,0,0,0,0,0,0,0};
  for (int k0 = 0; k0 < NP; k0 += 64) {
    __builtin_prefetch(arow0 + k0 + 256, 0, 1);
    __builtin_prefetch(brow0 + k0 + 256, 0, 1);
    union { v8i v; unsigned int u[8]; } a0, a1, b0, b1;
#pragma unroll
    for (int j = 0; j < 8; ++j) {
      const int ka = (j >> 1) * 16 + (j & 1) * 4 + (hi ? 8 : 0);
      const int kb = (j >> 1) * 16 + (j & 1) * 4 + (hi ? 8 : 0);
      a0.u[j] = *(const unsigned int*)(arow0 + k0 + ka);
      a1.u[j] = *(const unsigned int*)(arow1 + k0 + ka);
      b0.u[j] = *(const unsigned int*)(brow0 + k0 + kb);
      b1.u[j] = *(const unsigned int*)(brow1 + k0 + kb);
    }
    acc00 = __builtin_amdgcn_wmma_i32_16x16x64_iu8(false, a0.v, false, b0.v, acc00, false, false);
    acc01 = __builtin_amdgcn_wmma_i32_16x16x64_iu8(false, a0.v, false, b1.v, acc01, false, false);
    acc10 = __builtin_amdgcn_wmma_i32_16x16x64_iu8(false, a1.v, false, b0.v, acc10, false, false);
    acc11 = __builtin_amdgcn_wmma_i32_16x16x64_iu8(false, a1.v, false, b1.v, acc11, false, false);
  }
  v8i accs[4] = {acc00, acc01, acc10, acc11};
#pragma unroll
  for (int t = 0; t < 4; ++t) {
    const int mi = t >> 1, ni = t & 1;
    union { v8i v; int i[8]; } r; r.v = accs[t];
#pragma unroll
    for (int j = 0; j < 8; ++j) {
      const int row = tmm + mi * 16 + j + (hi ? 8 : 0);
      const int col = tnn + ni * 16 + mrow;
      const size_t o = base + (size_t)row * NP + col;
      unsigned char val;
      if (mode == 0) {
        val = (r.i[j] > 0) ? 1 : 0;
      } else {
        int s = r.i[j]; if (s > 255) s = 255;
        val = mask[o] ? (unsigned char)s : 0;
      }
      Out[o] = val;
    }
  }
}

__global__ __launch_bounds__(256) void u8f16_gemm_kernel(
    const unsigned char* __restrict__ Au8, const _Float16* __restrict__ Bf,
    _Float16* __restrict__ Out) {
  __shared__ _Float16     Bs[32 * DIM];
  __shared__ unsigned char As[16 * 32];
  const int wave = threadIdx.x >> 5;
  const int lane = threadIdx.x & 31;
  const int batch = blockIdx.y;
  const int tm = blockIdx.x * 16;
  const int tn = wave * 16;
  const int hi = lane >> 4, mrow = lane & 15;
  const int col = tn + mrow;
  const unsigned char* Abase = Au8 + (size_t)batch * NP * NP + (size_t)tm * NP;
  const _Float16* Bbase = Bf + (size_t)batch * NP * DIM;
  v8f acc = {0.f, 0.f, 0.f, 0.f, 0.f, 0.f, 0.f, 0.f};
  for (int k0 = 0; k0 < NP; k0 += 32) {
    __syncthreads();
    {
      const uint4* src = (const uint4*)(Bbase + (size_t)k0 * DIM);
#ifdef ASYNC_LDS
      __builtin_amdgcn_global_load_async_to_lds_b128(
          (gv4i*)(src + threadIdx.x),
          (lds_v4i_ptr)((char*)Bs + threadIdx.x * 16),
          0, 0);
      __builtin_amdgcn_global_load_async_to_lds_b128(
          (gv4i*)(src + 256 + threadIdx.x),
          (lds_v4i_ptr)((char*)Bs + 4096 + threadIdx.x * 16),
          0, 0);
#else
      uint4* dst = (uint4*)Bs;
      dst[threadIdx.x]       = src[threadIdx.x];
      dst[256 + threadIdx.x] = src[256 + threadIdx.x];
#endif
    }
    if (threadIdx.x < 128) {
      const int r = threadIdx.x >> 3, o = (threadIdx.x & 7) * 4;
      *(unsigned int*)(As + r * 32 + o) =
          *(const unsigned int*)(Abase + (size_t)r * NP + k0 + o);
    }
#ifdef ASYNC_LDS
    asm volatile("s_wait_asynccnt 0x0" ::: "memory");
#endif
    __syncthreads();
    union { v16h v; _Float16 h[16]; } a, b;
#pragma unroll
    for (int j = 0; j < 8; ++j) {
      const int ka = 2 * j + (hi ? 16 : 0);
      const int kb = 2 * j + (hi ? 16 : 0);
      a.h[2 * j]     = (_Float16)As[mrow * 32 + ka];
      a.h[2 * j + 1] = (_Float16)As[mrow * 32 + ka + 1];
      b.h[2 * j]     = Bs[(kb) * DIM + col];
      b.h[2 * j + 1] = Bs[(kb + 1) * DIM + col];
    }
    acc = __builtin_amdgcn_wmma_f32_16x16x32_f16(false, a.v, false, b.v,
                                                 (short)0, acc, false, false);
  }
  union { v8f v; float f[8]; } r; r.v = acc;
#pragma unroll
  for (int j = 0; j < 8; ++j) {
    const int row = tm + j + (hi ? 8 : 0);
    Out[((size_t)batch * NP + row) * DIM + col] = (_Float16)r.f[j];
  }
}

__global__ __launch_bounds__(256) void final_gemm_kernel(
    const _Float16* __restrict__ fv2h, const _Float16* __restrict__ W2h,
    const float* __restrict__ b2, float* __restrict__ out) {
  const int wave = threadIdx.x >> 5;
  const int lane = threadIdx.x & 31;
  const int batch = blockIdx.y;
  const int tile = blockIdx.x * 8 + wave;
  const int tm = (tile >> 3) * 16;
  const int tn = (tile & 7) * 16;
  const int hi = lane >> 4, mrow = lane & 15;
  const _Float16* arow = fv2h + ((size_t)batch * NP + tm + mrow) * DIM;
  const int col = tn + mrow;
  v8f acc = {0.f, 0.f, 0.f, 0.f, 0.f, 0.f, 0.f, 0.f};
  for (int k0 = 0; k0 < DIM; k0 += 32) {
    union { v16h v; _Float16 h[16]; } a, b;
#pragma unroll
    for (int j = 0; j < 8; ++j) {
      const int ka = 2 * j + (hi ? 16 : 0);
      const int kb = 2 * j + (hi ? 16 : 0);
      a.h[2 * j]     = arow[k0 + ka];
      a.h[2 * j + 1] = arow[k0 + ka + 1];
      b.h[2 * j]     = W2h[(size_t)(k0 + kb) * DIM + col];
      b.h[2 * j + 1] = W2h[(size_t)(k0 + kb + 1) * DIM + col];
    }
    acc = __builtin_amdgcn_wmma_f32_16x16x32_f16(false, a.v, false, b.v,
                                                 (short)0, acc, false, false);
  }
  const float bias = b2[col];
  union { v8f v; float f[8]; } r; r.v = acc;
#pragma unroll
  for (int j = 0; j < 8; ++j) {
    const int row = tm + j + (hi ? 8 : 0);
    if (row < NN)
      out[((size_t)batch * NN + row) * DIM + col] = fmaxf(r.f[j] + bias, 0.f);
  }
}

__global__ __launch_bounds__(128) void mean_kernel(
    const float* __restrict__ out, float* __restrict__ meanOut) {
  const int b = blockIdx.x, d = threadIdx.x;
  float s = 0.f;
  for (int n = 0; n < NN; ++n)
    s += out[((size_t)b * NN + n) * DIM + d];
  meanOut[b * DIM + d] = s * (1.0f / (float)NN);
}

extern "C" void kernel_launch(void* const* d_in, const int* in_sizes, int n_in,
                              void* d_out, int out_size, void* d_ws, size_t ws_size,
                              hipStream_t stream) {
  const float* loc      = (const float*)d_in[0];
  const float* deadline = (const float*)d_in[1];
  const float* depot    = (const float*)d_in[2];
  const float* W0       = (const float*)d_in[3];
  const float* b0       = (const float*)d_in[4];
  const float* W2       = (const float*)d_in[5];
  const float* b2       = (const float*)d_in[6];
  float* out = (float*)d_out;

  const size_t SZ_ADJ = (size_t)BATCH * NP * NP;
  const size_t SZ_FH  = (size_t)BATCH * NP * DIM * sizeof(_Float16);
  char* p = (char*)d_ws;
  unsigned char* A   = (unsigned char*)p; p += SZ_ADJ;
  unsigned char* B2m = (unsigned char*)p; p += SZ_ADJ;
  unsigned char* Cm  = (unsigned char*)p; p += SZ_ADJ;
  _Float16* fv0h = (_Float16*)p; p += SZ_FH;
  _Float16* fv1h = (_Float16*)p; p += SZ_FH;
  _Float16* fv2h = (_Float16*)p; p += SZ_FH;
  _Float16* W2h  = (_Float16*)p;

  adj_kernel<<<BATCH * NP, 256, 0, stream>>>(loc, depot, A);
  fv0_kernel<<<BATCH * NP, 128, 0, stream>>>(loc, deadline, depot, W0, b0, fv0h);
  cvt_kernel<<<(DIM * DIM + 255) / 256, 256, 0, stream>>>(W2, W2h);

  const int nmacro = NT32 * NT32;
  dim3 g_iu8((nmacro + 7) / 8, BATCH);
  iu8_gemm_kernel<<<g_iu8, 256, 0, stream>>>(A, A, nullptr, B2m, 0);

  iu8_gemm_kernel<<<g_iu8, 256, 0, stream>>>(B2m, A, B2m, Cm, 1);

  dim3 g_f16(NT16, BATCH);
  u8f16_gemm_kernel<<<g_f16, 256, 0, stream>>>(A, fv0h, fv1h);
  u8f16_gemm_kernel<<<g_f16, 256, 0, stream>>>(Cm, fv1h, fv2h);

  final_gemm_kernel<<<g_f16, 256, 0, stream>>>(fv2h, W2h, b2, out);
  mean_kernel<<<BATCH, 128, 0, stream>>>(out, out + (size_t)BATCH * NN * DIM);
}
